// MambaLayer_6382321401879
// MI455X (gfx1250) — compile-verified, matrix operand order checked by reading, not run on hardware
//
#include <hip/hip_runtime.h>

// ---------------------------------------------------------------------------
// Mamba layer for MI455X (gfx1250, wave32, WMMA bf16 16x16x32 + async->LDS)
// B=2, C(d_model)=384, L=16*16*16=4096, M=B*L=8192
// d_inner=768, d_state=16, dt_rank=24 (pad 32), d_conv=4
// ---------------------------------------------------------------------------

typedef __attribute__((ext_vector_type(16))) __bf16 v16bf;
typedef __attribute__((ext_vector_type(8)))  float  v8f;
typedef unsigned short u16;
typedef unsigned int   u32;
typedef unsigned long long u64;
typedef __attribute__((address_space(3))) u16 lds_u16;

#define BB   2
#define CM   384      // d_model
#define LL   4096     // tokens per batch
#define MM   8192     // B*L
#define DI   768      // d_inner
#define DS   16       // d_state
#define DTR  24       // dt_rank
#define DTRP 32       // padded dt_rank
#define NXP  64       // padded x_proj rows (56 -> 64)
#define TCH  64       // scan chunk length
#define NCH  (LL / TCH)

__device__ __forceinline__ u16 f2bf(float f) {
  u32 u = __float_as_uint(f);
  u32 r = u + 0x7FFFu + ((u >> 16) & 1u);   // round-to-nearest-even
  return (u16)(r >> 16);
}

// Load a 16x32 bf16 WMMA fragment from a row-major [16+, ld] u16 matrix at p0.
// Works for both global and LDS pointers. Per ISA 7.12.2 bf16 layouts.
__device__ __forceinline__ v16bf frag_from(const u16* p0, int ld) {
  int lane = threadIdx.x & 31;
  int half = lane >> 4;
  int r    = lane & 15;
  const u16* p = p0 + (size_t)r * ld;
  union { u32 u[8]; v16bf v; } f;
#pragma unroll
  for (int i = 0; i < 8; ++i) {
    int e = (i < 4) ? (half * 4 + i) : (8 + half * 4 + (i - 4));
    f.u[i] = *(const u32*)(p + 2 * e);
  }
  return f.v;
}

// ------------------------- weight conversion -------------------------------
__global__ void wconv_kernel(const float* __restrict__ in_proj_w,   // [1536,384]
                             const float* __restrict__ x_proj_w,    // [56,768]
                             const float* __restrict__ dt_proj_w,   // [768,24]
                             const float* __restrict__ out_proj_w,  // [384,768]
                             u16* __restrict__ w_in, u16* __restrict__ w_xp,
                             u16* __restrict__ w_dt, u16* __restrict__ w_out) {
  const int n_in  = 2 * DI * CM;
  const int n_xp  = NXP * DI;
  const int n_dt  = DI * DTRP;
  const int n_out = CM * DI;
  int total = n_in + n_xp + n_dt + n_out;
  for (int i = blockIdx.x * blockDim.x + threadIdx.x; i < total;
       i += gridDim.x * blockDim.x) {
    int j = i;
    if (j < n_in) { w_in[j] = f2bf(in_proj_w[j]); continue; }
    j -= n_in;
    if (j < n_xp) {
      int n = j / DI, k = j % DI;
      w_xp[j] = (n < DTR + 2 * DS) ? f2bf(x_proj_w[n * DI + k]) : (u16)0;
      continue;
    }
    j -= n_xp;
    if (j < n_dt) {
      int d = j / DTRP, k = j % DTRP;
      w_dt[j] = (k < DTR) ? f2bf(dt_proj_w[d * DTR + k]) : (u16)0;
      continue;
    }
    j -= n_dt;
    w_out[j] = f2bf(out_proj_w[j]);
  }
}

// ------------------------- layernorm -> bf16 -------------------------------
__global__ void ln_kernel(const float* __restrict__ x,     // [B, C, L]
                          const float* __restrict__ ln_w,
                          const float* __restrict__ ln_b,
                          u16* __restrict__ xn) {           // [M, C] bf16
  int m = blockIdx.x * blockDim.x + threadIdx.x;
  if (m >= MM) return;
  int b = m >> 12, l = m & (LL - 1);
  const float* px = x + (size_t)b * CM * LL + l;
  float s = 0.f, ss = 0.f;
  for (int c = 0; c < CM; ++c) { float v = px[(size_t)c * LL]; s += v; ss += v * v; }
  float mu  = s * (1.f / CM);
  float var = ss * (1.f / CM) - mu * mu;
  float rs  = rsqrtf(var + 1e-5f);
  u16* po = xn + (size_t)m * CM;
  for (int c = 0; c < CM; ++c) {
    float v = px[(size_t)c * LL];
    po[c] = f2bf((v - mu) * rs * ln_w[c] + ln_b[c]);
  }
}

// ---------------- WMMA GEMM: C[M,N] = A[M,K] * W[N,K]^T --------------------
// Block = 8 waves. Each wave computes a 16(M) x 64(N) strip: one A fragment
// feeds 4 WMMAs per k-step. The 64x32 bf16 weight slab for the block's N range
// is staged into LDS with GLOBAL_LOAD_ASYNC_TO_LDS_B128 (ASYNCcnt), shared by
// all 8 waves. TSTORE=1 scatters the output transposed into [B, C, L].
#define GEMM_WAVES 8
template <bool TSTORE>
__global__ void gemm_bf16_v2(const u16* __restrict__ A,
                             const u16* __restrict__ W,
                             float* __restrict__ Cout,
                             int N, int K) {
  __shared__ u16 sW[64 * 32];                       // 4 KB weight slab
  int wave   = threadIdx.x >> 5;
  int lane   = threadIdx.x & 31;
  int m_tile = blockIdx.y * GEMM_WAVES + wave;      // each wave: own M tile
  int n0     = blockIdx.x * 64;

  // per-thread async-copy slot: 256 threads x 16B = 4KB
  int row = threadIdx.x >> 2;                       // 0..63  (N row in slab)
  int quad = threadIdx.x & 3;                       // 0..3   (16B chunk in row)
  const u16* gW = W + (size_t)(n0 + row) * K + quad * 8;
  unsigned lds_dst = (unsigned)(u64)(lds_u16*)(sW + row * 32 + quad * 8);

  v8f c[4];
#pragma unroll
  for (int t = 0; t < 4; ++t) c[t] = (v8f){0.f,0.f,0.f,0.f,0.f,0.f,0.f,0.f};

  for (int k0 = 0; k0 < K; k0 += 32) {
    __syncthreads();                                // slab free to overwrite
    u64 ga = (u64)(gW + k0);
    asm volatile("global_load_async_to_lds_b128 %0, %1, off"
                 :: "v"(lds_dst), "v"(ga) : "memory");
    asm volatile("s_wait_asynccnt 0x0" ::: "memory");
    __syncthreads();                                // slab visible to all waves

    v16bf a = frag_from(A + (size_t)(m_tile * 16) * K + k0, K);
#pragma unroll
    for (int t = 0; t < 4; ++t) {
      v16bf b = frag_from(sW + t * 16 * 32, 32);
      c[t] = __builtin_amdgcn_wmma_f32_16x16x32_bf16(false, a, false, b,
                                                     (short)0, c[t],
                                                     false, false);
    }
  }

  int half = lane >> 4;
  int ncol = lane & 15;
#pragma unroll
  for (int t = 0; t < 4; ++t) {
    int n = n0 + t * 16 + ncol;
#pragma unroll
    for (int r = 0; r < 8; ++r) {
      int m = m_tile * 16 + r + 8 * half;
      if (TSTORE) {
        int b = m >> 12, l = m & (LL - 1);
        Cout[((size_t)b * CM + n) * LL + l] = c[t][r];
      } else {
        Cout[(size_t)m * N + n] = c[t][r];
      }
    }
  }
}

// ------------------------- causal depthwise conv + SiLU --------------------
__global__ void conv_silu_kernel(const float* __restrict__ xz,     // [M,1536]
                                 const float* __restrict__ conv_w, // [DI,4]
                                 const float* __restrict__ conv_b, // [DI]
                                 float* __restrict__ u_f32,        // [M,DI]
                                 u16*  __restrict__ u_bf16) {      // [M,DI]
  int idx = blockIdx.x * blockDim.x + threadIdx.x;
  if (idx >= MM * DI) return;
  int d = idx % DI, m = idx / DI;
  int l = m & (LL - 1), b = m >> 12;
  float acc = conv_b[d];
#pragma unroll
  for (int j = 0; j < 4; ++j) {
    int ll = l - 3 + j;
    if (ll >= 0)
      acc += conv_w[d * 4 + j] * xz[((size_t)(b * LL + ll)) * (2 * DI) + d];
  }
  float sv = acc * (1.f / (1.f + __expf(-acc)));   // SiLU
  u_f32[idx]  = sv;
  u_bf16[idx] = f2bf(sv);
}

// ------------------------- dt slice -> bf16 (pad 24->32) -------------------
__global__ void dt_prep_kernel(const float* __restrict__ xdbl,  // [M,NXP]
                               u16* __restrict__ dtlow) {       // [M,DTRP]
  int idx = blockIdx.x * blockDim.x + threadIdx.x;
  if (idx >= MM * DTRP) return;
  int m = idx / DTRP, k = idx % DTRP;
  dtlow[idx] = (k < DTR) ? f2bf(xdbl[(size_t)m * NXP + k]) : (u16)0;
}

// ------------------------- bias + softplus ---------------------------------
__global__ void softplus_kernel(float* __restrict__ delta,        // [M,DI]
                                const float* __restrict__ dt_b) { // [DI]
  int idx = blockIdx.x * blockDim.x + threadIdx.x;
  if (idx >= MM * DI) return;
  float v = delta[idx] + dt_b[idx % DI];
  delta[idx] = (v > 15.f) ? v : log1pf(__expf(v));
}

// ------------- selective scan: chunked parallel decomposition --------------
// Diagonal transition => exact chunk decomposition:
//   pass A: per-chunk local scan from 0, record S (final local state) and
//           P = prod(dA) over the chunk.            [parallel over B x NCH]
//   pass B: carry scan over chunk summaries -> H0.  [tiny, 1536 threads]
//   pass C: replay each chunk from H0, emit gated bf16 y. [parallel]
__global__ void scan_partial_kernel(const float* __restrict__ delta,  // [M,DI]
                                    const float* __restrict__ xdbl,   // [M,NXP]
                                    const float* __restrict__ u_f32,  // [M,DI]
                                    const float* __restrict__ A_log,  // [DI,DS]
                                    float* __restrict__ Pst,          // [B,NCH,DI,DS]
                                    float* __restrict__ Sst) {
  __shared__ float sB[TCH * DS];
  int b = blockIdx.y, ch = blockIdx.x, d = threadIdx.x;
  int l0 = ch * TCH;
  float A[DS];
#pragma unroll
  for (int n = 0; n < DS; ++n) A[n] = -__expf(A_log[d * DS + n]);
  for (int t = threadIdx.x; t < TCH * DS; t += blockDim.x) {
    int li = t >> 4, n = t & 15;
    sB[t] = xdbl[((size_t)(b * LL + l0 + li)) * NXP + DTR + n];
  }
  __syncthreads();
  float h[DS], P[DS];
#pragma unroll
  for (int n = 0; n < DS; ++n) { h[n] = 0.f; P[n] = 1.f; }
  for (int li = 0; li < TCH; ++li) {
    size_t m = (size_t)b * LL + l0 + li;
    float dl = delta[m * DI + d];
    float du = dl * u_f32[m * DI + d];
#pragma unroll
    for (int n = 0; n < DS; ++n) {
      float dA = __expf(dl * A[n]);
      h[n] = dA * h[n] + du * sB[li * DS + n];
      P[n] *= dA;
    }
  }
  size_t o = (((size_t)b * NCH + ch) * DI + d) * DS;
#pragma unroll
  for (int n = 0; n < DS; ++n) { Pst[o + n] = P[n]; Sst[o + n] = h[n]; }
}

__global__ void scan_carry_kernel(const float* __restrict__ Pst,
                                  const float* __restrict__ Sst,
                                  float* __restrict__ H0) {
  int idx = blockIdx.x * blockDim.x + threadIdx.x;   // over BB*DI
  if (idx >= BB * DI) return;
  int b = idx / DI, d = idx % DI;
  float h[DS];
#pragma unroll
  for (int n = 0; n < DS; ++n) h[n] = 0.f;
  for (int ch = 0; ch < NCH; ++ch) {
    size_t o = (((size_t)b * NCH + ch) * DI + d) * DS;
#pragma unroll
    for (int n = 0; n < DS; ++n) {
      H0[o + n] = h[n];                              // state entering chunk ch
      h[n] = Pst[o + n] * h[n] + Sst[o + n];
    }
  }
}

__global__ void scan_final_kernel(const float* __restrict__ delta,
                                  const float* __restrict__ xdbl,
                                  const float* __restrict__ u_f32,
                                  const float* __restrict__ xz,     // z @ col 768
                                  const float* __restrict__ A_log,
                                  const float* __restrict__ Dp,
                                  const float* __restrict__ H0,
                                  u16* __restrict__ y_bf16) {       // [M,DI]
  __shared__ float sB[TCH * DS];
  __shared__ float sC[TCH * DS];
  int b = blockIdx.y, ch = blockIdx.x, d = threadIdx.x;
  int l0 = ch * TCH;
  float A[DS];
#pragma unroll
  for (int n = 0; n < DS; ++n) A[n] = -__expf(A_log[d * DS + n]);
  float Dd = Dp[d];
  for (int t = threadIdx.x; t < TCH * 2 * DS; t += blockDim.x) {
    int li = t >> 5, n = t & 31;
    float v = xdbl[((size_t)(b * LL + l0 + li)) * NXP + DTR + n];
    if (n < DS) sB[li * DS + n] = v;
    else        sC[li * DS + (n - DS)] = v;
  }
  size_t o = (((size_t)b * NCH + ch) * DI + d) * DS;
  float h[DS];
#pragma unroll
  for (int n = 0; n < DS; ++n) h[n] = H0[o + n];
  __syncthreads();
  for (int li = 0; li < TCH; ++li) {
    size_t m = (size_t)b * LL + l0 + li;
    float dl = delta[m * DI + d];
    float uv = u_f32[m * DI + d];
    float du = dl * uv;
    float y = 0.f;
#pragma unroll
    for (int n = 0; n < DS; ++n) {
      float dA = __expf(dl * A[n]);
      h[n] = dA * h[n] + du * sB[li * DS + n];
      y += h[n] * sC[li * DS + n];
    }
    float zv = xz[m * (2 * DI) + DI + d];
    float g = zv * (1.f / (1.f + __expf(-zv)));      // SiLU(z)
    y_bf16[m * DI + d] = f2bf((y + uv * Dd) * g);
  }
}

// ---------------------------------------------------------------------------
extern "C" void kernel_launch(void* const* d_in, const int* in_sizes, int n_in,
                              void* d_out, int out_size, void* d_ws, size_t ws_size,
                              hipStream_t stream) {
  const float* x          = (const float*)d_in[0];
  const float* ln_w       = (const float*)d_in[1];
  const float* ln_b       = (const float*)d_in[2];
  const float* in_proj_w  = (const float*)d_in[3];
  const float* conv_w     = (const float*)d_in[4];
  const float* conv_b     = (const float*)d_in[5];
  const float* x_proj_w   = (const float*)d_in[6];
  const float* dt_proj_w  = (const float*)d_in[7];
  const float* dt_proj_b  = (const float*)d_in[8];
  const float* A_log      = (const float*)d_in[9];
  const float* Dp         = (const float*)d_in[10];
  const float* out_proj_w = (const float*)d_in[11];
  float* out = (float*)d_out;

  // ---- workspace carve-up (256B aligned) ----
  char* ws = (char*)d_ws;
  size_t off = 0;
  auto carve = [&](size_t bytes) -> void* {
    void* p = ws + off;
    off = (off + bytes + 255) & ~(size_t)255;
    return p;
  };
  u16*   xn    = (u16*)  carve((size_t)MM * CM * 2);
  u16*   w_in  = (u16*)  carve((size_t)2 * DI * CM * 2);
  u16*   w_xp  = (u16*)  carve((size_t)NXP * DI * 2);
  u16*   w_dt  = (u16*)  carve((size_t)DI * DTRP * 2);
  u16*   w_out = (u16*)  carve((size_t)CM * DI * 2);
  float* xz    = (float*)carve((size_t)MM * 2 * DI * 4);
  float* u32f  = (float*)carve((size_t)MM * DI * 4);
  u16*   ub16  = (u16*)  carve((size_t)MM * DI * 2);
  float* xdbl  = (float*)carve((size_t)MM * NXP * 4);
  u16*   dtlow = (u16*)  carve((size_t)MM * DTRP * 2);
  float* delta = (float*)carve((size_t)MM * DI * 4);
  u16*   ybf   = (u16*)  carve((size_t)MM * DI * 2);
  const size_t nsum = (size_t)BB * NCH * DI * DS;
  float* Pst = (float*)carve(nsum * 4);
  float* Sst = (float*)carve(nsum * 4);
  float* H0  = (float*)carve(nsum * 4);
  (void)ws_size; (void)in_sizes; (void)n_in; (void)out_size;

  // 1. weights -> bf16 (with padding)
  wconv_kernel<<<512, 256, 0, stream>>>(in_proj_w, x_proj_w, dt_proj_w,
                                        out_proj_w, w_in, w_xp, w_dt, w_out);
  // 2. layernorm
  ln_kernel<<<MM / 256, 256, 0, stream>>>(x, ln_w, ln_b, xn);
  // 3. in_proj: [8192,384] x [1536,384]^T -> [8192,1536]
  gemm_bf16_v2<false><<<dim3((2 * DI) / 64, MM / 16 / GEMM_WAVES),
                        32 * GEMM_WAVES, 0, stream>>>(xn, w_in, xz, 2 * DI, CM);
  // 4. depthwise causal conv + SiLU
  conv_silu_kernel<<<(MM * DI) / 256, 256, 0, stream>>>(xz, conv_w, conv_b,
                                                        u32f, ub16);
  // 5. x_proj: [8192,768] x [64,768]^T -> [8192,64]
  gemm_bf16_v2<false><<<dim3(NXP / 64, MM / 16 / GEMM_WAVES),
                        32 * GEMM_WAVES, 0, stream>>>(ub16, w_xp, xdbl, NXP, DI);
  // 6. dt slice -> bf16 padded
  dt_prep_kernel<<<(MM * DTRP) / 256, 256, 0, stream>>>(xdbl, dtlow);
  // 7. dt_proj: [8192,32] x [768,32]^T -> [8192,768]; then bias+softplus
  gemm_bf16_v2<false><<<dim3(DI / 64, MM / 16 / GEMM_WAVES),
                        32 * GEMM_WAVES, 0, stream>>>(dtlow, w_dt, delta, DI, DTRP);
  softplus_kernel<<<(MM * DI) / 256, 256, 0, stream>>>(delta, dt_proj_b);
  // 8. selective scan (chunk-parallel) + D-skip + SiLU gate -> y bf16
  scan_partial_kernel<<<dim3(NCH, BB), DI, 0, stream>>>(delta, xdbl, u32f,
                                                        A_log, Pst, Sst);
  scan_carry_kernel<<<(BB * DI + 255) / 256, 256, 0, stream>>>(Pst, Sst, H0);
  scan_final_kernel<<<dim3(NCH, BB), DI, 0, stream>>>(delta, xdbl, u32f, xz,
                                                      A_log, Dp, H0, ybf);
  // 9. out_proj with transposed store to [B, C, L]
  gemm_bf16_v2<true><<<dim3(CM / 64, MM / 16 / GEMM_WAVES),
                       32 * GEMM_WAVES, 0, stream>>>(ybf, w_out, out, CM, DI);
}
